// StatefulAttentionLSTM_29016799051871
// MI455X (gfx1250) — matrix operand order repaired and checked by reading, compiled, not run on hardware
//
#include <hip/hip_runtime.h>
#include <hip/hip_bf16.h>
#include <math.h>

#define BB 64
#define TT 1024
#define II 64
#define DD 256
#define GG (4 * DD)      // 1024 gate outputs
#define KC (II + DD)     // 320 fused K dim for LSTM gemm
#define KT 32            // attention key tile

typedef __attribute__((ext_vector_type(16))) _Float16 v16h;
typedef __attribute__((ext_vector_type(8)))  _Float16 v8h;
typedef __attribute__((ext_vector_type(8)))  float    v8f;

__device__ __forceinline__ v8f wmma16(v16h a, v16h b, v8f c) {
  // 8 args: (neg_a, A, neg_b, B, c_mod, C, reuse_a, reuse_b)
  return __builtin_amdgcn_wmma_f32_16x16x32_f16(false, a, false, b, (short)0, c,
                                                false, false);
}

// Opaque zero offset in an SGPR. Adding it to a base pointer keeps pointer
// provenance (so InferAddressSpaces still emits global_load, not flat_load)
// while making the address loop-variant, so LICM cannot hoist (and then
// spill) large fragment sets out of hot loops.
__device__ __forceinline__ int opaque_zero() {
  int z = 0;
  asm volatile("" : "+s"(z));
  return z;
}

// A fragment (16x32 f16) from a row-major matrix: rows 0..15 at `base`,
// `ld` halves per row (multiple of 8), K = k0..k0+31.
// ISA layout: lanes 0-15 hold K=[k0,k0+8) in halves 0-7 and K=[k0+16,k0+24)
// in halves 8-15; lanes 16-31 are shifted by +8 in K.
__device__ __forceinline__ v16h load_a(const _Float16* base, int ld, int k0, int lane) {
  const int m  = lane & 15;
  const int hi = (lane >> 4) & 1;
  const _Float16* p = base + (size_t)m * ld + k0 + (hi << 3);
  v8h x0 = *(const v8h*)(p);
  v8h x1 = *(const v8h*)(p + 16);
  return __builtin_shufflevector(x0, x1, 0, 1, 2, 3, 4, 5, 6, 7,
                                 8, 9, 10, 11, 12, 13, 14, 15);
}

// B fragment (32x16 f16) for computing A @ W^T with W row-major [N][K]:
// B(k,n) = W[n0+n][k0+k]. ISA layout: lane holds fixed n = lane%16, K=[8h,8h+8)
// in halves 0-7 and K=[16+8h,24+8h) in halves 8-15 (h = lane/16).
__device__ __forceinline__ v16h load_bT(const _Float16* W, int ld, int n0, int k0, int lane) {
  const int n  = lane & 15;
  const int kg = (lane >> 4) << 3;
  const _Float16* p = W + (size_t)(n0 + n) * ld + k0 + kg;
  v8h x0 = *(const v8h*)(p);
  v8h x1 = *(const v8h*)(p + 16);
  return __builtin_shufflevector(x0, x1, 0, 1, 2, 3, 4, 5, 6, 7,
                                 8, 9, 10, 11, 12, 13, 14, 15);
}

__device__ __forceinline__ float sigm(float x) { return 1.0f / (1.0f + __expf(-x)); }

// ---------------- prep kernels ----------------
__global__ void k_cvt(const float* __restrict__ src, _Float16* __restrict__ dst, int n) {
  int stride = gridDim.x * blockDim.x;
  for (int i = blockIdx.x * blockDim.x + threadIdx.x; i < n; i += stride)
    dst[i] = (_Float16)src[i];
}

__global__ void k_pack(const float* __restrict__ Wih, const float* __restrict__ Whh,
                       const float* __restrict__ bih, const float* __restrict__ bhh,
                       _Float16* __restrict__ Wcat, float* __restrict__ bsum) {
  int stride = gridDim.x * blockDim.x;
  for (int i = blockIdx.x * blockDim.x + threadIdx.x; i < GG * KC; i += stride) {
    int r = i / KC, c = i - r * KC;
    float v = (c < II) ? Wih[r * II + c] : Whh[r * DD + (c - II)];
    Wcat[i] = (_Float16)v;
  }
  for (int i = blockIdx.x * blockDim.x + threadIdx.x; i < GG; i += stride)
    bsum[i] = bih[i] + bhh[i];
}

// ---------------- LSTM recurrence ----------------
// grid = B/16 workgroups; workgroup owns 16 batches. Per step:
// gates[16,1024] = [x_t | h] (LDS, 16x320) @ Wcat^T (global f16, L2-resident).
// Wave w owns gate columns d in [32w, 32w+32): for each d it holds i/f/g/o
// (tiles g*16 + 2w + jj), so the state update is register-local; c stays in
// v8f accumulators for the whole scan.
__launch_bounds__(256, 1)
__global__ void k_lstm(const _Float16* __restrict__ Xh,    // [B,T,I] f16
                       const _Float16* __restrict__ Wcat,  // [G,KC]  f16
                       const float* __restrict__ bsum,     // [G]
                       _Float16* __restrict__ Hh,          // [B,T,D] f16
                       _Float16* __restrict__ comb) {      // [B*T,2D] f16
  __shared__ _Float16 Abuf[16][KC + 8];  // 16 x 328 halves
  const int tid  = threadIdx.x;
  const int lane = tid & 31;
  const int w    = tid >> 5;
  const int b0   = blockIdx.x * 16;

  for (int i = tid; i < 16 * (KC + 8); i += 256)
    (&Abuf[0][0])[i] = (_Float16)0.0f;

  v8f cacc[2];
#pragma unroll
  for (int jj = 0; jj < 2; ++jj)
#pragma unroll
    for (int e = 0; e < 8; ++e) cacc[jj][e] = 0.0f;
  __syncthreads();

  for (int t = 0; t < TT; ++t) {
    // stage x_t: 16 rows x 64 halves
    if (tid < 128) {
      int row = tid >> 3;
      int col = (tid & 7) * 8;
      const _Float16* src = Xh + ((size_t)(b0 + row) * TT + t) * II + col;
      *(v8h*)&Abuf[row][col] = *(const v8h*)src;
    }
    __syncthreads();

    v8f acc[4][2];
#pragma unroll
    for (int g = 0; g < 4; ++g)
#pragma unroll
      for (int jj = 0; jj < 2; ++jj) {
        float bv = bsum[g * 256 + w * 32 + jj * 16 + (lane & 15)];
#pragma unroll
        for (int e = 0; e < 8; ++e) acc[g][jj][e] = bv;
      }

    // Loop-variant zero offset: B-fragments must be streamed from L2 every
    // step, never hoisted into (spilled) registers. Provenance preserved ->
    // these stay global_load_b128, not flat_load_b128.
    const _Float16* Wc = Wcat + opaque_zero();
    for (int k = 0; k < KC; k += 32) {
      v16h a = load_a(&Abuf[0][0], KC + 8, k, lane);
#pragma unroll
      for (int g = 0; g < 4; ++g)
#pragma unroll
        for (int jj = 0; jj < 2; ++jj) {
          v16h bf = load_bT(Wc, KC, g * 256 + w * 32 + jj * 16, k, lane);
          acc[g][jj] = wmma16(a, bf, acc[g][jj]);
        }
    }
    __syncthreads();  // all waves done reading h-region of Abuf

    // activations + state update + write-back (torch gate order i,f,g,o)
#pragma unroll
    for (int jj = 0; jj < 2; ++jj) {
#pragma unroll
      for (int e = 0; e < 8; ++e) {
        float ig = sigm(acc[0][jj][e]);
        float fg = sigm(acc[1][jj][e]);
        float gg = tanhf(acc[2][jj][e]);
        float og = sigm(acc[3][jj][e]);
        float c  = fg * cacc[jj][e] + ig * gg;
        cacc[jj][e] = c;
        float h = og * tanhf(c);
        int m = e + ((lane >> 4) << 3);
        int d = w * 32 + jj * 16 + (lane & 15);
        _Float16 hv = (_Float16)h;
        Abuf[m][II + d] = hv;
        size_t row = (size_t)(b0 + m) * TT + t;
        Hh[row * DD + d]         = hv;
        comb[row * (2 * DD) + d] = hv;
      }
    }
    __syncthreads();
  }
}

// ---------------- causal flash attention (Q=K=V=H, unscaled) ----------------
// grid = (T/128, B); wave w owns query rows [qr0, qr0+16). Streams 32-row
// K/V tiles through LDS (V stored transposed so PV B-fragments are contiguous).
__launch_bounds__(256, 1)
__global__ void k_attn(const _Float16* __restrict__ Hh, _Float16* __restrict__ comb) {
  __shared__ _Float16 Ks[KT][DD + 8];       // 32 x 264
  __shared__ _Float16 Vt[DD][KT + 8];       // 256 x 40 (transposed V)
  __shared__ _Float16 Pb[8][16][KT + 8];    // per-wave P tile, 16 x 40
  const int tid  = threadIdx.x;
  const int lane = tid & 31;
  const int w    = tid >> 5;
  const int b    = blockIdx.y;
  const int q0   = blockIdx.x * 128;
  const int qr0  = q0 + w * 16;
  // Wave-uniform query-tile index, laundered into an SGPR so the Q base
  // pointer is provably uniform.
  const int qr0u = __builtin_amdgcn_readfirstlane(qr0);
  const _Float16* Hb = Hh + (size_t)b * TT * DD;
  const _Float16* Qbase = Hb + (size_t)qr0u * DD;

  v8f oacc[16];
#pragma unroll
  for (int nt = 0; nt < 16; ++nt)
#pragma unroll
    for (int e = 0; e < 8; ++e) oacc[nt][e] = 0.0f;
  float rmax[8], rsum[8];
#pragma unroll
  for (int r = 0; r < 8; ++r) { rmax[r] = -1e30f; rsum[r] = 0.0f; }

  const int smax_wg = q0 + 127;
  for (int s0 = 0; s0 <= smax_wg; s0 += KT) {
    // cooperative staging: 32 rows x 256 halves, also transposed into Vt
    {
      int row   = tid >> 3;
      int cbase = (tid & 7) * 32;
      const _Float16* src = Hb + (size_t)(s0 + row) * DD + cbase;
#pragma unroll
      for (int u = 0; u < 4; ++u) {
        v8h v = *(const v8h*)(src + u * 8);
        *(v8h*)&Ks[row][cbase + u * 8] = v;
#pragma unroll
        for (int e = 0; e < 8; ++e) Vt[cbase + u * 8 + e][row] = v[e];
      }
    }
    __syncthreads();

    if (__builtin_amdgcn_readfirstlane(s0 <= qr0u + 15)) {
      // Q fragments re-read from cache each tile; do NOT pin 64 VGPRs of Q
      // across the whole key loop on top of the 128-VGPR O accumulator.
      const _Float16* Qb = Qbase + opaque_zero();
      // S = Q @ K^T for this 16x32 score tile
      v8f sacc[2];
#pragma unroll
      for (int hh = 0; hh < 2; ++hh)
#pragma unroll
        for (int e = 0; e < 8; ++e) sacc[hh][e] = 0.0f;
      for (int k = 0; k < DD; k += 32) {
        v16h a = load_a(Qb, DD, k, lane);
        sacc[0] = wmma16(a, load_bT(&Ks[0][0], DD + 8, 0, k, lane), sacc[0]);
        sacc[1] = wmma16(a, load_bT(&Ks[0][0], DD + 8, 16, k, lane), sacc[1]);
      }

      // causal mask + online softmax (row m spread across 16 lanes)
#pragma unroll
      for (int r = 0; r < 8; ++r) {
        int m = r + ((lane >> 4) << 3);
        int q = qr0u + m;
        float v0 = sacc[0][r], v1 = sacc[1][r];
        int c0 = s0 + (lane & 15);
        if (c0 > q) v0 = -1e30f;
        if (c0 + 16 > q) v1 = -1e30f;
        float smx = fmaxf(v0, v1);
#pragma unroll
        for (int off = 8; off >= 1; off >>= 1)
          smx = fmaxf(smx, __shfl_xor(smx, off, 32));
        float mnew  = fmaxf(rmax[r], smx);
        float scale = __expf(rmax[r] - mnew);
        float p0 = __expf(v0 - mnew);
        float p1 = __expf(v1 - mnew);
        float ps = p0 + p1;
#pragma unroll
        for (int off = 8; off >= 1; off >>= 1)
          ps += __shfl_xor(ps, off, 32);
        rsum[r] = rsum[r] * scale + ps;
        rmax[r] = mnew;
        Pb[w][m][(lane & 15)]      = (_Float16)p0;
        Pb[w][m][16 + (lane & 15)] = (_Float16)p1;
#pragma unroll
        for (int nt = 0; nt < 16; ++nt) oacc[nt][r] *= scale;
      }

      // O += P @ V  (single K=32 step; same-wave LDS ops are in-order)
      v16h pa = load_a(&Pb[w][0][0], KT + 8, 0, lane);
#pragma unroll
      for (int nt = 0; nt < 16; ++nt) {
        v16h bf = load_bT(&Vt[0][0], KT + 8, nt * 16, 0, lane);
        oacc[nt] = wmma16(pa, bf, oacc[nt]);
      }
    }
    __syncthreads();
  }

  // normalize and store context into comb[..., D:2D] as f16
#pragma unroll
  for (int nt = 0; nt < 16; ++nt)
#pragma unroll
    for (int r = 0; r < 8; ++r) {
      int m = r + ((lane >> 4) << 3);
      size_t row = (size_t)b * TT + qr0u + m;
      float v = oacc[nt][r] / rsum[r];
      comb[row * (2 * DD) + DD + nt * 16 + (lane & 15)] = (_Float16)v;
    }
}

// ---------------- fused MLP: out = relu(comb @ W1^T + b1) @ W2^T + b2 ----------------
__launch_bounds__(256, 1)
__global__ void k_mlp(const _Float16* __restrict__ comb,  // [M, 2D]
                      const _Float16* __restrict__ W1h,   // [D, 2D]
                      const float* __restrict__ b1,
                      const _Float16* __restrict__ W2h,   // [I, D]
                      const float* __restrict__ b2,
                      float* __restrict__ out) {          // [M, I]
  __shared__ _Float16 Hid[8][16][DD + 8];  // per-wave 16 x 264
  const int tid  = threadIdx.x;
  const int lane = tid & 31;
  const int w    = tid >> 5;
  const size_t mr0 = (size_t)blockIdx.x * 128 + (size_t)w * 16;

  v8f acc[16];
#pragma unroll
  for (int nt = 0; nt < 16; ++nt) {
    float bv = b1[nt * 16 + (lane & 15)];
#pragma unroll
    for (int e = 0; e < 8; ++e) acc[nt][e] = bv;
  }
  for (int k = 0; k < 2 * DD; k += 32) {
    v16h a = load_a(comb + mr0 * (2 * DD), 2 * DD, k, lane);
#pragma unroll
    for (int nt = 0; nt < 16; ++nt)
      acc[nt] = wmma16(a, load_bT(W1h, 2 * DD, nt * 16, k, lane), acc[nt]);
  }
  // relu -> f16 -> wave-private LDS tile (same-wave LDS is in-order, no barrier)
#pragma unroll
  for (int nt = 0; nt < 16; ++nt)
#pragma unroll
    for (int r = 0; r < 8; ++r) {
      int m = r + ((lane >> 4) << 3);
      Hid[w][m][nt * 16 + (lane & 15)] = (_Float16)fmaxf(acc[nt][r], 0.0f);
    }

  v8f acc2[4];
#pragma unroll
  for (int nt = 0; nt < 4; ++nt) {
    float bv = b2[nt * 16 + (lane & 15)];
#pragma unroll
    for (int e = 0; e < 8; ++e) acc2[nt][e] = bv;
  }
  for (int k = 0; k < DD; k += 32) {
    v16h a = load_a(&Hid[w][0][0], DD + 8, k, lane);
#pragma unroll
    for (int nt = 0; nt < 4; ++nt)
      acc2[nt] = wmma16(a, load_bT(W2h, DD, nt * 16, k, lane), acc2[nt]);
  }
#pragma unroll
  for (int nt = 0; nt < 4; ++nt)
#pragma unroll
    for (int r = 0; r < 8; ++r) {
      int m = r + ((lane >> 4) << 3);
      out[(mr0 + m) * II + nt * 16 + (lane & 15)] = acc2[nt][r];
    }
}

extern "C" void kernel_launch(void* const* d_in, const int* in_sizes, int n_in,
                              void* d_out, int out_size, void* d_ws, size_t ws_size,
                              hipStream_t stream) {
  const float* x   = (const float*)d_in[0];
  const float* Wih = (const float*)d_in[1];
  const float* Whh = (const float*)d_in[2];
  const float* bih = (const float*)d_in[3];
  const float* bhh = (const float*)d_in[4];
  const float* W1  = (const float*)d_in[5];
  const float* b1  = (const float*)d_in[6];
  const float* W2  = (const float*)d_in[7];
  const float* b2  = (const float*)d_in[8];
  float* out = (float*)d_out;

  char* ws = (char*)d_ws;
  size_t off = 0;
  auto alloc = [&](size_t bytes) -> void* {
    void* p = ws + off;
    off += (bytes + 255) & ~(size_t)255;
    return p;
  };
  _Float16* Hh   = (_Float16*)alloc((size_t)BB * TT * DD * 2);        // 32 MB
  _Float16* comb = (_Float16*)alloc((size_t)BB * TT * 2 * DD * 2);    // 64 MB
  _Float16* Xh   = (_Float16*)alloc((size_t)BB * TT * II * 2);        // 8 MB
  _Float16* Wcat = (_Float16*)alloc((size_t)GG * KC * 2);
  float*    bsum = (float*)alloc((size_t)GG * 4);
  _Float16* W1h  = (_Float16*)alloc((size_t)DD * 2 * DD * 2);
  _Float16* W2h  = (_Float16*)alloc((size_t)II * DD * 2);

  k_cvt<<<1024, 256, 0, stream>>>(x, Xh, BB * TT * II);
  k_cvt<<<256, 256, 0, stream>>>(W1, W1h, DD * 2 * DD);
  k_cvt<<<64, 256, 0, stream>>>(W2, W2h, II * DD);
  k_pack<<<512, 256, 0, stream>>>(Wih, Whh, bih, bhh, Wcat, bsum);

  k_lstm<<<BB / 16, 256, 0, stream>>>(Xh, Wcat, bsum, Hh, comb);

  dim3 ag(TT / 128, BB);
  k_attn<<<ag, 256, 0, stream>>>(Hh, comb);

  k_mlp<<<(BB * TT) / 128, 256, 0, stream>>>(comb, W1h, b1, W2h, b2, out);
}
